// CausalSelfAttention_39213051412481
// MI455X (gfx1250) — matrix operand order repaired and checked by reading, not compiled, not run on hardware
//
#include <hip/hip_runtime.h>
#include <hip/hip_bf16.h>
#include <stdint.h>

// ---- CDNA5 wave32 WMMA types ------------------------------------------------
typedef __bf16        v16bf __attribute__((ext_vector_type(16)));
typedef float         v8f   __attribute__((ext_vector_type(8)));
typedef unsigned int  uv4   __attribute__((ext_vector_type(4)));
typedef float         fv4   __attribute__((ext_vector_type(4)));

#define S_LEN 8192
#define D_IN  2048
#define D_HD  128

union Frag16 { v16bf v; uv4 q[2]; unsigned int d[8]; unsigned short h[16]; };
union FragC  { v8f   v; float f[8]; };
union PK4    { uv4 q; unsigned int d[4]; };

// ---- bf16 pack (RNE) --------------------------------------------------------
#if __has_builtin(__builtin_amdgcn_cvt_pk_bf16_f32)
__device__ __forceinline__ unsigned pack2(float lo, float hi) {
  auto t = __builtin_amdgcn_cvt_pk_bf16_f32(lo, hi);
  return __builtin_bit_cast(unsigned, t);
}
#else
__device__ __forceinline__ unsigned pack2(float lo, float hi) {
  unsigned ul = __builtin_bit_cast(unsigned, lo);
  unsigned uh = __builtin_bit_cast(unsigned, hi);
  unsigned rl = (ul + 0x7FFFu + ((ul >> 16) & 1u)) >> 16;
  unsigned rh = (uh + 0x7FFFu + ((uh >> 16) & 1u)) >> 16;
  return rl | (rh << 16);
}
#endif
__device__ __forceinline__ unsigned short f2bf(float x) {
  return (unsigned short)(pack2(x, x) & 0xffffu);
}

// ---- pipelined GEMM helpers -------------------------------------------------
__device__ __forceinline__ void load_b8(Frag16* dst, const unsigned short* wbase,
                                        int k0, int lm, int half) {
#pragma unroll
  for (int f = 0; f < 8; ++f) {
    const unsigned short* wp = wbase + (size_t)(f * 16 + lm) * D_IN + k0 + half * 8;
    dst[f].q[0] = *(const uv4*)(wp);
    dst[f].q[1] = *(const uv4*)(wp + 16);
  }
}
__device__ __forceinline__ void load_a_raw(fv4* ar, const float* xrow, int k0, int half) {
  ar[0] = *(const fv4*)(xrow + k0 + half * 8);
  ar[1] = *(const fv4*)(xrow + k0 + half * 8 + 4);
  ar[2] = *(const fv4*)(xrow + k0 + 16 + half * 8);
  ar[3] = *(const fv4*)(xrow + k0 + 16 + half * 8 + 4);
}
__device__ __forceinline__ Frag16 cvt_a(const fv4* ar) {
  Frag16 a;
  a.d[0] = pack2(ar[0].x, ar[0].y); a.d[1] = pack2(ar[0].z, ar[0].w);
  a.d[2] = pack2(ar[1].x, ar[1].y); a.d[3] = pack2(ar[1].z, ar[1].w);
  a.d[4] = pack2(ar[2].x, ar[2].y); a.d[5] = pack2(ar[2].z, ar[2].w);
  a.d[6] = pack2(ar[3].x, ar[3].y); a.d[7] = pack2(ar[3].z, ar[3].w);
  return a;
}
__device__ __forceinline__ void do_wmma8(FragC* acc, const Frag16& a, const Frag16* b) {
#pragma unroll
  for (int f = 0; f < 8; ++f)
    acc[f].v = __builtin_amdgcn_wmma_f32_16x16x32_bf16(
        false, a.v, false, b[f].v, (short)0, acc[f].v, false, false);
}

// ---- Kernel 1: weights fp32 [2048,128] -> bf16 transposed [128,2048] --------
// w_q additionally scaled by log2(e)/sqrt(128) so softmax can use exp2.
__global__ __launch_bounds__(256) void wconv_kernel(const float* __restrict__ wq,
                                                    const float* __restrict__ wk,
                                                    const float* __restrict__ wv,
                                                    unsigned short* __restrict__ wT) {
  int idx = blockIdx.x * 256 + threadIdx.x;       // 3*2048*128 = 786432 threads
  int mat = idx >> 18;                            // / 262144
  int rem = idx & 262143;
  int k = rem >> 7;
  int n = rem & 127;
  const float* w = (mat == 0) ? wq : (mat == 1) ? wk : wv;
  float s = (mat == 0) ? (1.4426950408889634f * 0.0883883476483184f) : 1.0f;
  wT[mat * (D_HD * D_IN) + n * D_IN + k] = f2bf(w[k * D_HD + n] * s);
}

// ---- Kernel 2: Q/K/V projection, software-pipelined bf16 WMMA GEMM ----------
// grid.x = 3 * 64; block = 256 (8 waves). Wave computes a 16x128 output tile.
// Two-deep ping-pong on the K dimension so loads overlap WMMAs.
__global__ __launch_bounds__(256) void proj_kernel(const float* __restrict__ x,
                                                   const unsigned short* __restrict__ wT,
                                                   unsigned short* __restrict__ Qb,
                                                   unsigned short* __restrict__ Kb,
                                                   unsigned short* __restrict__ Vt) {
  const int lane = threadIdx.x & 31, wave = threadIdx.x >> 5;
  const int half = lane >> 4, lm = lane & 15;
  const int mat  = blockIdx.x >> 6;
  const int row0 = (((int)blockIdx.x & 63) * 8 + wave) * 16;
  const unsigned short* wbase = wT + mat * (D_HD * D_IN);

  FragC acc[8];
#pragma unroll
  for (int f = 0; f < 8; ++f)
#pragma unroll
    for (int r = 0; r < 8; ++r) acc[f].f[r] = 0.f;

  const float* xrow = x + (size_t)(row0 + lm) * D_IN;

  Frag16 b0[8], b1[8];
  fv4 a0r[4], a1r[4];
  load_b8(b0, wbase, 0, lm, half);
  load_a_raw(a0r, xrow, 0, half);

  for (int kc2 = 0; kc2 < 32; ++kc2) {           // two 32-chunks per iteration
    const int k0 = kc2 * 64;
    __builtin_prefetch(xrow + k0 + 256, 0, 0);   // stream x ahead (global_prefetch_b8)

    // issue loads for chunk k0+32 before computing chunk k0
    load_b8(b1, wbase, k0 + 32, lm, half);
    load_a_raw(a1r, xrow, k0 + 32, half);
    {
      Frag16 a = cvt_a(a0r);
      do_wmma8(acc, a, b0);
    }
    // issue loads for chunk k0+64 before computing chunk k0+32
    if (kc2 < 31) {
      load_b8(b0, wbase, k0 + 64, lm, half);
      load_a_raw(a0r, xrow, k0 + 64, half);
    }
    {
      Frag16 a = cvt_a(a1r);
      do_wmma8(acc, a, b1);
    }
  }

  if (mat < 2) {
    unsigned short* outm = (mat == 0) ? Qb : Kb;
#pragma unroll
    for (int f = 0; f < 8; ++f)
#pragma unroll
      for (int r = 0; r < 8; ++r)
        outm[(size_t)(row0 + half * 8 + r) * D_HD + f * 16 + lm] = f2bf(acc[f].f[r]);
  } else {
    // 8 consecutive seq positions per lane are contiguous in Vt -> one b128 store
#pragma unroll
    for (int f = 0; f < 8; ++f) {
      PK4 p;
      p.d[0] = pack2(acc[f].f[0], acc[f].f[1]);
      p.d[1] = pack2(acc[f].f[2], acc[f].f[3]);
      p.d[2] = pack2(acc[f].f[4], acc[f].f[5]);
      p.d[3] = pack2(acc[f].f[6], acc[f].f[7]);
      *(uv4*)(Vt + (size_t)(f * 16 + lm) * S_LEN + row0 + half * 8) = p.q;
    }
  }
}

// ---- Kernel 3: causal flash attention, one 16-query tile per wave -----------
// grid.x = 128; block = 128 (4 waves -> 512 waves total, one tile each).
__global__ __launch_bounds__(128) void attn_kernel(const unsigned short* __restrict__ Qb,
                                                   const unsigned short* __restrict__ Kb,
                                                   const unsigned short* __restrict__ Vt,
                                                   float* __restrict__ out) {
  __shared__ unsigned short pstage[4][16 * 32];   // per-wave P staging (1 KB each)
  const int lane = threadIdx.x & 31, wave = threadIdx.x >> 5;
  const int half = lane >> 4, lm = lane & 15;
  const int q0 = ((int)blockIdx.x * 4 + wave) * 16;

  // Q fragments for this tile (16 x 128 bf16, already scaled by log2e/sqrt(dk))
  Frag16 qf[4];
  const unsigned short* qrow = Qb + (size_t)(q0 + lm) * D_HD;
#pragma unroll
  for (int c = 0; c < 4; ++c) {
    qf[c].q[0] = *(const uv4*)(qrow + c * 32 + half * 16);
    qf[c].q[1] = *(const uv4*)(qrow + c * 32 + half * 16 + 8);
  }

  FragC o[8];
#pragma unroll
  for (int f = 0; f < 8; ++f)
#pragma unroll
    for (int r = 0; r < 8; ++r) o[f].f[r] = 0.f;
  float m_run[8], l_run[8];
#pragma unroll
  for (int r = 0; r < 8; ++r) { m_run[r] = -3.0e38f; l_run[r] = 0.f; }

  unsigned short* st = &pstage[wave][0];

  for (int j0 = 0; j0 < q0 + 16; j0 += 32) {
    // prefetch next key block of K and V^T
    __builtin_prefetch(Kb + (size_t)(j0 + 32 + lm) * D_HD, 0, 0);
    __builtin_prefetch(Vt + (size_t)(lm * 16) * S_LEN + j0 + 32, 0, 0);

    // ---- issue all K-fragment loads (2 key tiles x 4 dkq chunks) ----
    Frag16 kb[8];
#pragma unroll
    for (int c = 0; c < 4; ++c) {
      const unsigned short* kp0 = Kb + (size_t)(j0 + lm) * D_HD + c * 32 + half * 16;
      const unsigned short* kp1 = Kb + (size_t)(j0 + 16 + lm) * D_HD + c * 32 + half * 16;
      kb[c].q[0]     = *(const uv4*)kp0; kb[c].q[1]     = *(const uv4*)(kp0 + 8);
      kb[4 + c].q[0] = *(const uv4*)kp1; kb[4 + c].q[1] = *(const uv4*)(kp1 + 8);
    }
    // ---- issue all V-fragment loads NOW; they drain during scores+softmax ----
    Frag16 vb[8];
#pragma unroll
    for (int f = 0; f < 8; ++f) {
      const unsigned short* vp = Vt + (size_t)(f * 16 + lm) * S_LEN + j0 + half * 16;
      vb[f].q[0] = *(const uv4*)vp; vb[f].q[1] = *(const uv4*)(vp + 8);
    }

    // ---- S = Q K^T (8 WMMAs) ----
    FragC c0, c1;
#pragma unroll
    for (int r = 0; r < 8; ++r) { c0.f[r] = 0.f; c1.f[r] = 0.f; }
#pragma unroll
    for (int c = 0; c < 4; ++c) {
      c0.v = __builtin_amdgcn_wmma_f32_16x16x32_bf16(
          false, qf[c].v, false, kb[c].v, (short)0, c0.v, false, false);
      c1.v = __builtin_amdgcn_wmma_f32_16x16x32_bf16(
          false, qf[c].v, false, kb[4 + c].v, (short)0, c1.v, false, false);
    }

    // ---- causal mask + online softmax (exp2 domain) ----
    float p0[8], p1[8], alpha[8];
#pragma unroll
    for (int r = 0; r < 8; ++r) {
      const int qg = q0 + half * 8 + r;
      float s0 = ((j0 + lm) <= qg) ? c0.f[r] : -3.0e38f;
      float s1 = ((j0 + 16 + lm) <= qg) ? c1.f[r] : -3.0e38f;
      float tm = fmaxf(s0, s1);
      tm = fmaxf(tm, __shfl_xor(tm, 1, 32));
      tm = fmaxf(tm, __shfl_xor(tm, 2, 32));
      tm = fmaxf(tm, __shfl_xor(tm, 4, 32));
      tm = fmaxf(tm, __shfl_xor(tm, 8, 32));
      const float mn = fmaxf(m_run[r], tm);
      alpha[r] = __builtin_amdgcn_exp2f(m_run[r] - mn);
      p0[r] = __builtin_amdgcn_exp2f(s0 - mn);
      p1[r] = __builtin_amdgcn_exp2f(s1 - mn);
      float rs = p0[r] + p1[r];
      rs += __shfl_xor(rs, 1, 32);
      rs += __shfl_xor(rs, 2, 32);
      rs += __shfl_xor(rs, 4, 32);
      rs += __shfl_xor(rs, 8, 32);
      l_run[r] = alpha[r] * l_run[r] + rs;
      m_run[r] = mn;
    }
#pragma unroll
    for (int f = 0; f < 8; ++f)
#pragma unroll
      for (int r = 0; r < 8; ++r) o[f].f[r] *= alpha[r];

    // ---- transpose P (C-layout f32) -> A-layout bf16 via per-wave LDS ----
#pragma unroll
    for (int r = 0; r < 8; ++r) {
      const int row = half * 8 + r;
      st[row * 32 + lm]      = f2bf(p0[r]);
      st[row * 32 + 16 + lm] = f2bf(p1[r]);
    }
    // same-wave DS ops are in-order; wait + compiler barrier before reload
    __asm volatile("s_wait_dscnt 0" ::: "memory");
    Frag16 pf;
    pf.q[0] = *(const uv4*)(st + lm * 32 + half * 16);
    pf.q[1] = *(const uv4*)(st + lm * 32 + half * 16 + 8);
    __asm volatile("" ::: "memory");

    // ---- O += P V (8 WMMAs over dv=128); V already resident ----
#pragma unroll
    for (int f = 0; f < 8; ++f)
      o[f].v = __builtin_amdgcn_wmma_f32_16x16x32_bf16(
          false, pf.v, false, vb[f].v, (short)0, o[f].v, false, false);
  }

  // ---- normalize and store fp32 output ----
#pragma unroll
  for (int f = 0; f < 8; ++f)
#pragma unroll
    for (int r = 0; r < 8; ++r) {
      const int row = q0 + half * 8 + r;
      out[(size_t)row * D_HD + f * 16 + lm] = o[f].f[r] / l_run[r];
    }
}

// ---- host launcher ----------------------------------------------------------
extern "C" void kernel_launch(void* const* d_in, const int* in_sizes, int n_in,
                              void* d_out, int out_size, void* d_ws, size_t ws_size,
                              hipStream_t stream) {
  (void)in_sizes; (void)n_in; (void)out_size; (void)ws_size;
  const float* x  = (const float*)d_in[0];
  const float* wq = (const float*)d_in[1];
  const float* wk = (const float*)d_in[2];
  const float* wv = (const float*)d_in[3];
  float* out = (float*)d_out;

  // workspace: wT (3*128*2048) | Q (8192*128) | K (8192*128) | Vt (128*8192), all bf16
  unsigned short* wT = (unsigned short*)d_ws;
  unsigned short* Qb = wT + 3 * D_HD * D_IN;
  unsigned short* Kb = Qb + (size_t)S_LEN * D_HD;
  unsigned short* Vt = Kb + (size_t)S_LEN * D_HD;

  wconv_kernel<<<(3 * D_IN * D_HD) / 256, 256, 0, stream>>>(wq, wk, wv, wT);
  proj_kernel<<<3 * 64, 256, 0, stream>>>(x, wT, Qb, Kb, Vt);
  attn_kernel<<<128, 128, 0, stream>>>(Qb, Kb, Vt, out);
}
